// GAEModel_19035295056030
// MI455X (gfx1250) — matrix operand order repaired and checked by reading, not compiled, not run on hardware
//
#include <hip/hip_runtime.h>
#include <hip/hip_bf16.h>
#include <math.h>

typedef __attribute__((ext_vector_type(16))) __bf16 v16bf;
typedef __attribute__((ext_vector_type(8)))  float  v8f;

#define N_NODES 50000
#define N_EDGES 800000
#define C_IN    128
#define C_HID   128
#define C_OUT   64

// ---------------- degree / normalization ----------------
__global__ void init_deg_kernel(float* deg) {
    int i = blockIdx.x * blockDim.x + threadIdx.x;
    if (i < N_NODES) deg[i] = 1.0f;  // self-loop
}

__global__ void count_deg_kernel(const int* __restrict__ dst, float* deg) {
    int e = blockIdx.x * blockDim.x + threadIdx.x;
    if (e < N_EDGES) atomicAdd(&deg[dst[e]], 1.0f);
}

__global__ void finalize_deg_kernel(float* deg) {
    int i = blockIdx.x * blockDim.x + threadIdx.x;
    if (i < N_NODES) deg[i] = rsqrtf(deg[i]);  // buffer now holds deg^-1/2
}

// ---------------- weight pack: fp32 [K,N] (optionally transposed src) -> bf16 fragment layout --
// Bpack element index: (k/32)*N*32 + n*32 + (k%32)
// Per-lane WMMA B fragment (col n, half h, k-step k0) = 16 contiguous bf16 at
//   ((k0/32)*N + n)*32 + h*16   -> one 32B vector load.
__global__ void pack_weight_kernel(const float* __restrict__ W,
                                   __bf16* __restrict__ Bpack,
                                   int K, int N, int transpose) {
    int i = blockIdx.x * blockDim.x + threadIdx.x;
    if (i >= K * N) return;
    int k = i / N;
    int n = i % N;
    float v = transpose ? W[n * K + k] : W[k * N + n];
    Bpack[((k >> 5) * N + n) * 32 + (k & 31)] = (__bf16)v;
}

// ---------------- WMMA GEMM: C[M,N] = (relu?)A[M,K] @ B[K,N], bf16 in / f32 acc ----------------
// grid.x = M/16 row tiles; blockDim.x = 32 * (N/16); wave w owns column tile w.
template <int K, int N, bool RELU>
__global__ void wmma_gemm_kernel(const float* __restrict__ A,
                                 const __bf16* __restrict__ Bpack,
                                 float* __restrict__ C) {
    const int lane = threadIdx.x & 31;
    const int wave = threadIdx.x >> 5;
    const int half = lane >> 4;       // lane group 0..15 / 16..31
    const int l16  = lane & 15;
    const int m0   = blockIdx.x * 16;
    const int n0   = wave * 16;
    const int col  = n0 + l16;

    const float* arow = A + (long)(m0 + l16) * K + half * 16;

    v8f acc = {};
    #pragma unroll
    for (int k0 = 0; k0 < K; k0 += 32) {
        // A fragment: element e -> K = e + 16*half
        // -> one contiguous 16-float chunk at (k0 + 16*half)
        const float4 f0 = *reinterpret_cast<const float4*>(arow + k0);
        const float4 f1 = *reinterpret_cast<const float4*>(arow + k0 + 4);
        const float4 f2 = *reinterpret_cast<const float4*>(arow + k0 + 8);
        const float4 f3 = *reinterpret_cast<const float4*>(arow + k0 + 12);
        float av[16] = {f0.x, f0.y, f0.z, f0.w, f1.x, f1.y, f1.z, f1.w,
                        f2.x, f2.y, f2.z, f2.w, f3.x, f3.y, f3.z, f3.w};
        v16bf a;
        #pragma unroll
        for (int e = 0; e < 16; ++e) {
            float v = av[e];
            if (RELU) v = fmaxf(v, 0.0f);
            a[e] = (__bf16)v;
        }
        // B fragment: 16 contiguous bf16 (32B aligned) in packed layout
        const v16bf b = *reinterpret_cast<const v16bf*>(
            Bpack + ((long)(k0 >> 5) * N + col) * 32 + half * 16);

        acc = __builtin_amdgcn_wmma_f32_16x16x32_bf16(
                  false, a, false, b, (short)0, acc, false, false);
    }
    // D layout: VGPR r -> row m0 + r + 8*half, col = n0 + l16
    float* crow = C + (long)(m0 + half * 8) * N + col;
    #pragma unroll
    for (int r = 0; r < 8; ++r)
        crow[(long)r * N] = acc[r];
}

// ---------------- aggregation init: agg = tmp * dinv^2 + bias (self-loop + bias) ----------------
__global__ void init_agg_kernel(const float* __restrict__ tmp,
                                const float* __restrict__ dinv,
                                const float* __restrict__ bias,
                                float* __restrict__ agg, int Cc) {
    long i = (long)blockIdx.x * blockDim.x + threadIdx.x;
    if (i >= (long)N_NODES * Cc) return;
    int n = (int)(i / Cc);
    int c = (int)(i % Cc);
    float di = dinv[n];
    agg[i] = tmp[i] * di * di + bias[c];
}

// ---------------- edge scatter: agg[dst] += tmp[src] * dinv[src]*dinv[dst] ----------------
// Cc/4 threads per edge; lanes within a wave cover contiguous float4 chunks -> coalesced
__global__ void scatter_kernel(const float* __restrict__ tmp,
                               const int* __restrict__ src,
                               const int* __restrict__ dst,
                               const float* __restrict__ dinv,
                               float* __restrict__ agg, int Cc) {
    const int per = Cc >> 2;
    long tid = (long)blockIdx.x * blockDim.x + threadIdx.x;
    if (tid >= (long)N_EDGES * per) return;
    int e  = (int)(tid / per);
    int c4 = (int)(tid % per) << 2;
    int s = src[e], d = dst[e];
    float norm = dinv[s] * dinv[d];
    const float4 v = *reinterpret_cast<const float4*>(tmp + (long)s * Cc + c4);
    float* out = agg + (long)d * Cc + c4;
    atomicAdd(out + 0, v.x * norm);
    atomicAdd(out + 1, v.y * norm);
    atomicAdd(out + 2, v.z * norm);
    atomicAdd(out + 3, v.w * norm);
}

// ---------------- decoder: out[e] = sigmoid(dot(z[src], p[dst]) + bb), wave per edge ----------------
__global__ void decode_kernel(const float* __restrict__ z,
                              const float* __restrict__ p,
                              const int* __restrict__ src,
                              const int* __restrict__ dst,
                              const float* __restrict__ bb,
                              float* __restrict__ out) {
    long gid = (long)blockIdx.x * blockDim.x + threadIdx.x;
    int wid  = (int)(gid >> 5);
    int lane = (int)(gid & 31);
    if (wid >= N_EDGES) return;
    int s = src[wid], d = dst[wid];
    const float* zs = z + (long)s * C_OUT;
    const float* pd = p + (long)d * C_OUT;
    float acc = zs[lane] * pd[lane] + zs[lane + 32] * pd[lane + 32];
    #pragma unroll
    for (int off = 16; off > 0; off >>= 1)
        acc += __shfl_down(acc, off, 32);
    if (lane == 0) {
        float t = acc + bb[0];
        out[wid] = 1.0f / (1.0f + __expf(-t));
    }
}

extern "C" void kernel_launch(void* const* d_in, const int* in_sizes, int n_in,
                              void* d_out, int out_size, void* d_ws, size_t ws_size,
                              hipStream_t stream) {
    const float* x    = (const float*)d_in[0];
    const int*   ei   = (const int*)d_in[1];      // edge_index [2, E]
    const int*   srcI = ei;
    const int*   dstI = ei + N_EDGES;
    const float* W1   = (const float*)d_in[2];
    const float* b1   = (const float*)d_in[3];
    const float* W2   = (const float*)d_in[4];
    const float* b2   = (const float*)d_in[5];
    const float* Wb   = (const float*)d_in[6];
    const float* bb   = (const float*)d_in[7];
    float* out = (float*)d_out;

    // workspace layout (all 256B aligned)
    char* ws = (char*)d_ws;
    size_t off = 0;
    auto alloc = [&](size_t bytes) -> void* {
        void* ptr = ws + off;
        off = (off + bytes + 255) & ~(size_t)255;
        return ptr;
    };
    float*  dinv = (float*)alloc((size_t)N_NODES * 4);
    float*  bufA = (float*)alloc((size_t)N_NODES * C_HID * 4);  // tmp1, later p
    float*  bufB = (float*)alloc((size_t)N_NODES * C_HID * 4);  // agg1 (=h1 pre-relu)
    float*  bufC = (float*)alloc((size_t)N_NODES * C_OUT * 4);  // tmp2
    float*  bufD = (float*)alloc((size_t)N_NODES * C_OUT * 4);  // z
    __bf16* W1p  = (__bf16*)alloc((size_t)C_IN * C_HID * 2);
    __bf16* W2p  = (__bf16*)alloc((size_t)C_HID * C_OUT * 2);
    __bf16* WbTp = (__bf16*)alloc((size_t)C_OUT * C_OUT * 2);

    const int rowTiles = N_NODES / 16;  // 3125 (exact)

    // normalization terms
    init_deg_kernel<<<(N_NODES + 255) / 256, 256, 0, stream>>>(dinv);
    count_deg_kernel<<<(N_EDGES + 255) / 256, 256, 0, stream>>>(dstI, dinv);
    finalize_deg_kernel<<<(N_NODES + 255) / 256, 256, 0, stream>>>(dinv);

    // bf16 weight packing (fragment-native layout); Wb packed transposed
    pack_weight_kernel<<<(C_IN * C_HID + 255) / 256, 256, 0, stream>>>(
        W1, W1p, C_IN, C_HID, 0);
    pack_weight_kernel<<<(C_HID * C_OUT + 255) / 256, 256, 0, stream>>>(
        W2, W2p, C_HID, C_OUT, 0);
    pack_weight_kernel<<<(C_OUT * C_OUT + 255) / 256, 256, 0, stream>>>(
        Wb, WbTp, C_OUT, C_OUT, 1);

    // layer 1: tmp1 = x @ W1
    wmma_gemm_kernel<C_IN, C_HID, false><<<rowTiles, 32 * (C_HID / 16), 0, stream>>>(
        x, W1p, bufA);
    init_agg_kernel<<<(int)(((long)N_NODES * C_HID + 255) / 256), 256, 0, stream>>>(
        bufA, dinv, b1, bufB, C_HID);
    scatter_kernel<<<(int)(((long)N_EDGES * (C_HID / 4) + 255) / 256), 256, 0, stream>>>(
        bufA, srcI, dstI, dinv, bufB, C_HID);

    // layer 2: tmp2 = relu(h1) @ W2
    wmma_gemm_kernel<C_HID, C_OUT, true><<<rowTiles, 32 * (C_OUT / 16), 0, stream>>>(
        bufB, W2p, bufC);
    init_agg_kernel<<<(int)(((long)N_NODES * C_OUT + 255) / 256), 256, 0, stream>>>(
        bufC, dinv, b2, bufD, C_OUT);
    scatter_kernel<<<(int)(((long)N_EDGES * (C_OUT / 4) + 255) / 256), 256, 0, stream>>>(
        bufC, srcI, dstI, dinv, bufD, C_OUT);

    // decoder precompute: p = z @ Wb^T
    wmma_gemm_kernel<C_OUT, C_OUT, false><<<rowTiles, 32 * (C_OUT / 16), 0, stream>>>(
        bufD, WbTp, bufA);

    // per-edge sigmoid(dot(z[src], p[dst]) + bb)
    decode_kernel<<<(int)(((long)N_EDGES * 32 + 255) / 256), 256, 0, stream>>>(
        bufD, bufA, srcI, dstI, bb, out);
}
